// GPSLayer_3667902070849
// MI455X (gfx1250) — matrix operand order repaired and checked by reading, not compiled, not run on hardware
//
#include <hip/hip_runtime.h>

// ---------------------------------------------------------------------------
// Types for CDNA5 WMMA (wave32): v_wmma_f32_16x16x32_bf16
// ---------------------------------------------------------------------------
typedef __attribute__((ext_vector_type(16))) __bf16 v16bf;
typedef __attribute__((ext_vector_type(8)))  float  v8f;

union Frag {
    uint4 q[2];   // 32 bytes
    v16bf v;      // 16 x bf16
};

static __device__ inline v8f wmma_bf16(v16bf a, v16bf b, v8f c) {
    // 8 args: (neg_a, A, neg_b, B, c_mod, C, reuse_a, reuse_b)
    return __builtin_amdgcn_wmma_f32_16x16x32_bf16(false, a, false, b, (short)0, c,
                                                   false, false);
}

static __device__ inline unsigned short f2bf(float f) {
    union { float f; unsigned u; } v; v.f = f;
    unsigned u = v.u;
    unsigned r = u + 0x7FFFu + ((u >> 16) & 1u);   // round-to-nearest-even
    return (unsigned short)(r >> 16);
}

// CDNA5 async memory->LDS copy (16B per lane), tracked by ASYNCcnt.
// VDST operand = VGPR holding per-lane LDS byte address; VADDR = 64-bit global.
static __device__ inline void async_b128(unsigned lds_off, const void* gaddr) {
    asm volatile("global_load_async_to_lds_b128 %0, %1, off"
                 :: "v"(lds_off), "v"(gaddr) : "memory");
}
static __device__ inline void wait_async0() {
    asm volatile("s_wait_asynccnt 0x0" ::: "memory");
}
static __device__ inline unsigned lds_off_u32(const void* p) {
    return (unsigned)(size_t)p;   // generic shared pointer: low 32 bits = LDS offset
}

// ---------------------------------------------------------------------------
// f32 -> bf16 conversion (plain and transposed)
// ---------------------------------------------------------------------------
__global__ void cvt_bf16_k(const float* __restrict__ src,
                           unsigned short* __restrict__ dst, long long n) {
    long long i = (long long)blockIdx.x * blockDim.x + threadIdx.x;
    long long stride = (long long)gridDim.x * blockDim.x;
    for (; i < n; i += stride) dst[i] = f2bf(src[i]);
}

// src[K][Nc] (row-major f32) -> dst[Nc][K] (row-major bf16)
__global__ void cvt_t_bf16_k(const float* __restrict__ src,
                             unsigned short* __restrict__ dst, int K, int Nc) {
    long long n = (long long)K * Nc;
    long long i = (long long)blockIdx.x * blockDim.x + threadIdx.x;
    long long stride = (long long)gridDim.x * blockDim.x;
    for (; i < n; i += stride) {
        int k = (int)(i / Nc), c = (int)(i % Nc);
        dst[(size_t)c * K + k] = f2bf(src[i]);
    }
}

__global__ void fill_f32_k(float* __restrict__ p, float v, long long n) {
    long long i = (long long)blockIdx.x * blockDim.x + threadIdx.x;
    long long stride = (long long)gridDim.x * blockDim.x;
    for (; i < n; i += stride) p[i] = v;
}

// ---------------------------------------------------------------------------
// GCN degree / dinv / scatter
// ---------------------------------------------------------------------------
__global__ void deg_count_k(const long long* __restrict__ eidx, long long E,
                            float* __restrict__ deg) {
    long long e = (long long)blockIdx.x * blockDim.x + threadIdx.x;
    if (e < E) atomicAdd(&deg[eidx[E + e]], 1.0f);   // dst row
}

__global__ void dinv_k(const float* __restrict__ deg, float* __restrict__ dinv, int n) {
    int i = blockIdx.x * blockDim.x + threadIdx.x;
    if (i < n) { float d = deg[i]; dinv[i] = d > 0.f ? rsqrtf(d) : 0.f; }
}

// one block (256 threads = feature dim) per edge
__global__ __launch_bounds__(256)
void scatter_k(const long long* __restrict__ eidx, long long E,
               const float* __restrict__ dinv, const float* __restrict__ xw,
               float* __restrict__ agg) {
    long long e = blockIdx.x;
    int c = threadIdx.x;
    long long s = eidx[e];
    long long d = eidx[E + e];
    float w = dinv[s] * dinv[d];
    atomicAdd(&agg[d * 256 + c], xw[s * 256 + c] * w);
}

// ---------------------------------------------------------------------------
// Block-wide LayerNorm helpers (block = 256 threads = one row, D = 256)
// ---------------------------------------------------------------------------
static __device__ inline float block_sum256(float v, float* sbuf) {
    for (int off = 16; off; off >>= 1) v += __shfl_xor(v, off, 32);
    if ((threadIdx.x & 31) == 0) sbuf[threadIdx.x >> 5] = v;
    __syncthreads();
    float t = 0.f;
    for (int j = 0; j < 8; ++j) t += sbuf[j];
    __syncthreads();
    return t;
}

// x1 = LN(x + agg + dinv^2*xw + b_gcn)
__global__ __launch_bounds__(256)
void gcn_ln_k(const float* __restrict__ x, const float* __restrict__ agg,
              const float* __restrict__ xw, const float* __restrict__ dinv,
              const float* __restrict__ bg, const float* __restrict__ g,
              const float* __restrict__ b, float* __restrict__ out,
              unsigned short* __restrict__ outbf) {
    __shared__ float sbuf[8];
    size_t i = blockIdx.x;
    int c = threadIdx.x;
    float di = dinv[i];
    float v = x[i * 256 + c] + agg[i * 256 + c] + di * di * xw[i * 256 + c] + bg[c];
    float mu = block_sum256(v, sbuf) * (1.0f / 256.0f);
    float xc = v - mu;
    float var = block_sum256(xc * xc, sbuf) * (1.0f / 256.0f);
    float inv = rsqrtf(var + 1e-5f);
    float o = xc * inv * g[c] + b[c];
    out[i * 256 + c] = o;
    if (outbf) outbf[i * 256 + c] = f2bf(o);
}

// out = LN(xres + y)
__global__ __launch_bounds__(256)
void add_ln_k(const float* __restrict__ xres, const float* __restrict__ y,
              const float* __restrict__ g, const float* __restrict__ b,
              float* __restrict__ out, unsigned short* __restrict__ outbf) {
    __shared__ float sbuf[8];
    size_t i = blockIdx.x;
    int c = threadIdx.x;
    float v = xres[i * 256 + c] + y[i * 256 + c];
    float mu = block_sum256(v, sbuf) * (1.0f / 256.0f);
    float xc = v - mu;
    float var = block_sum256(xc * xc, sbuf) * (1.0f / 256.0f);
    float inv = rsqrtf(var + 1e-5f);
    float o = xc * inv * g[c] + b[c];
    out[i * 256 + c] = o;
    if (outbf) outbf[i * 256 + c] = f2bf(o);
}

// ---------------------------------------------------------------------------
// Tiled bf16 WMMA GEMM:  C[M,Nc] = A[M,K] @ B[K,Nc] (+bias, relu)
// B supplied PRE-TRANSPOSED: Bt[Nc][K].  Block tile 128x128, 8 waves
// (4 row x 2 col), each wave 32x64 (2x4 WMMA tiles).  Tiles staged with
// global_load_async_to_lds_b128 (ASYNCcnt).
// If tN > 0, the bf16 output is written transposed: Cbf[col*tN + row].
// ---------------------------------------------------------------------------
__global__ __launch_bounds__(256)
void gemm_bf16_k(const unsigned short* __restrict__ A,
                 const unsigned short* __restrict__ Bt,
                 const float* __restrict__ bias,
                 float* __restrict__ C, unsigned short* __restrict__ Cbf,
                 int M, int Nc, int K, int relu, int tN) {
    __shared__ unsigned short Alds[128 * 40];   // [row][k], pad 40
    __shared__ unsigned short Btl [128 * 40];   // [col][k], pad 40
    const int tid  = threadIdx.x;
    const int lane = tid & 31;
    const int wave = tid >> 5;
    const int wr = wave >> 1;            // 0..3 (32-row strip)
    const int wc = wave & 1;             // 0..1 (64-col strip)
    const int half = lane >> 4;
    const int l15  = lane & 15;
    const int rowbase = blockIdx.y * 128;
    const int colbase = blockIdx.x * 128;

    v8f acc[2][4];
    for (int i = 0; i < 2; ++i)
        for (int j = 0; j < 4; ++j)
            for (int e = 0; e < 8; ++e) acc[i][j][e] = 0.f;

    const int r    = tid >> 1;   // 0..127: doubles as A-row and Bt-col index
    const int part = tid & 1;    // 16-element half of a 32-element k-slice

    for (int k0 = 0; k0 < K; k0 += 32) {
        // async-stage A tile 128x32 and Bt tile 128x32 (16B per issue per lane)
        {
            const unsigned short* ga = A  + (size_t)(rowbase + r) * K + k0 + part * 16;
            unsigned aoff = lds_off_u32(&Alds[r * 40 + part * 16]);
            async_b128(aoff,      ga);
            async_b128(aoff + 16, ga + 8);
            const unsigned short* gb = Bt + (size_t)(colbase + r) * K + k0 + part * 16;
            unsigned boff = lds_off_u32(&Btl[r * 40 + part * 16]);
            async_b128(boff,      gb);
            async_b128(boff + 16, gb + 8);
        }
        if (k0 + 32 < K) {
            __builtin_prefetch(A  + (size_t)(rowbase + r) * K + k0 + 32, 0, 1);
            __builtin_prefetch(Bt + (size_t)(colbase + r) * K + k0 + 32, 0, 1);
        }
        wait_async0();
        __syncthreads();

        Frag a[2], bfr[4];
        for (int mt = 0; mt < 2; ++mt) {
            int row = wr * 32 + mt * 16 + l15;
            a[mt].q[0] = *(const uint4*)&Alds[row * 40 + half * 8];        // K {0..7}+8h
            a[mt].q[1] = *(const uint4*)&Alds[row * 40 + 16 + half * 8];   // K {16..23}+8h
        }
        for (int nt = 0; nt < 4; ++nt) {
            int col = wc * 64 + nt * 16 + l15;
            bfr[nt].q[0] = *(const uint4*)&Btl[col * 40 + half * 8];       // K {0..7}+8h
            bfr[nt].q[1] = *(const uint4*)&Btl[col * 40 + 16 + half * 8];
        }
        for (int mt = 0; mt < 2; ++mt)
            for (int nt = 0; nt < 4; ++nt)
                acc[mt][nt] = wmma_bf16(a[mt].v, bfr[nt].v, acc[mt][nt]);
        __syncthreads();
    }

    // epilogue: C layout row = r + 8*half, col = l15 within each 16x16 tile
    for (int mt = 0; mt < 2; ++mt)
        for (int nt = 0; nt < 4; ++nt) {
            int col = colbase + wc * 64 + nt * 16 + l15;
            float bv = bias ? bias[col] : 0.0f;
            for (int rr = 0; rr < 8; ++rr) {
                int row = rowbase + wr * 32 + mt * 16 + rr + half * 8;
                float v = acc[mt][nt][rr] + bv;
                if (relu) v = fmaxf(v, 0.0f);
                if (C) C[(size_t)row * Nc + col] = v;
                if (Cbf) {
                    size_t idx = tN ? ((size_t)col * tN + row)
                                    : ((size_t)row * Nc + col);
                    Cbf[idx] = f2bf(v);
                }
            }
        }
}

// ---------------------------------------------------------------------------
// Flash attention, head dim 64, query tile 64 (4 waves x 16 rows), key tile 64
// V supplied PRE-TRANSPOSED: Vt_global[256][N].  grid = (N/64, H), block = 128
// ---------------------------------------------------------------------------
__global__ __launch_bounds__(128)
void flash_k(const unsigned short* __restrict__ Q,
             const unsigned short* __restrict__ Kb,
             const unsigned short* __restrict__ VtG,
             unsigned short* __restrict__ O, int N) {
    __shared__ unsigned short Klds[64 * 72];       // [key][dh]
    __shared__ unsigned short Vt  [64 * 72];       // [dh][key]
    __shared__ unsigned short Plds[4 * 16 * 72];   // per-wave P tile [row][key]
    const int tid  = threadIdx.x;
    const int lane = tid & 31;
    const int wave = tid >> 5;        // 0..3
    const int half = lane >> 4;
    const int l15  = lane & 15;
    const int h  = blockIdx.y;
    const int q0 = blockIdx.x * 64;
    const int D  = 256;
    const int co = h * 64;
    const float sl2 = 0.125f * 1.44269504088896f;  // 1/sqrt(64) * log2(e)

    // Q fragments for this wave's 16 rows (A-matrix layout, K split at 32)
    Frag qf[2];
    {
        const unsigned short* qrow = Q + (size_t)(q0 + wave * 16 + l15) * D + co;
        for (int kk = 0; kk < 2; ++kk) {
            qf[kk].q[0] = *(const uint4*)(qrow + kk * 32 + half * 8);
            qf[kk].q[1] = *(const uint4*)(qrow + kk * 32 + 16 + half * 8);
        }
    }

    v8f o_acc[4];
    for (int dt = 0; dt < 4; ++dt)
        for (int r = 0; r < 8; ++r) o_acc[dt][r] = 0.f;
    float m_run[8], l_run[8];
    for (int r = 0; r < 8; ++r) { m_run[r] = -1e30f; l_run[r] = 0.f; }

    unsigned short* Pw = &Plds[wave * 16 * 72];
    const int sr   = tid >> 1;   // 0..63: key row for K, dh row for Vt
    const int part = tid & 1;    // 32-element half of a 64-element row

    for (int kt = 0; kt < N; kt += 64) {
        // async-stage K tile [key][dh] and Vt tile [dh][key]
        {
            const unsigned short* gk = Kb + (size_t)(kt + sr) * D + co + part * 32;
            unsigned koff = lds_off_u32(&Klds[sr * 72 + part * 32]);
            const unsigned short* gv = VtG + (size_t)(co + sr) * N + kt + part * 32;
            unsigned voff = lds_off_u32(&Vt[sr * 72 + part * 32]);
            for (int i2 = 0; i2 < 4; ++i2) {
                async_b128(koff + i2 * 16, gk + i2 * 8);
                async_b128(voff + i2 * 16, gv + i2 * 8);
            }
        }
        wait_async0();
        __syncthreads();

        // S = Q K^T  (B-frag = K rows read contiguously)
        v8f s[4];
        for (int nt = 0; nt < 4; ++nt)
            for (int r = 0; r < 8; ++r) s[nt][r] = 0.f;
        for (int kk = 0; kk < 2; ++kk) {
            Frag bf[4];
            for (int nt = 0; nt < 4; ++nt) {
                const unsigned short* kr = &Klds[(nt * 16 + l15) * 72 + kk * 32 + half * 8];
                bf[nt].q[0] = *(const uint4*)kr;
                bf[nt].q[1] = *(const uint4*)(kr + 16);
            }
            for (int nt = 0; nt < 4; ++nt)
                s[nt] = wmma_bf16(qf[kk].v, bf[nt].v, s[nt]);
        }

        // online softmax (row = r + 8*half; 16 lanes per row)
        for (int r = 0; r < 8; ++r) {
            float mx = -1e30f;
            for (int nt = 0; nt < 4; ++nt) mx = fmaxf(mx, s[nt][r]);
            for (int off = 1; off < 16; off <<= 1) mx = fmaxf(mx, __shfl_xor(mx, off, 32));
            mx *= sl2;
            float mn = fmaxf(m_run[r], mx);
            float corr = exp2f(m_run[r] - mn);
            float rs = 0.f;
            int prow = (r + half * 8) * 72;
            for (int nt = 0; nt < 4; ++nt) {
                float p = exp2f(s[nt][r] * sl2 - mn);
                Pw[prow + nt * 16 + l15] = f2bf(p);
                rs += p;
            }
            for (int off = 1; off < 16; off <<= 1) rs += __shfl_xor(rs, off, 32);
            l_run[r] = l_run[r] * corr + rs;
            m_run[r] = mn;
            for (int dt = 0; dt < 4; ++dt) o_acc[dt][r] *= corr;
        }
        __syncthreads();

        // O += P V  (A-frag from Plds, B-frag from Vt rows, contiguous)
        for (int kk = 0; kk < 2; ++kk) {
            Frag pa;
            const unsigned short* pr = &Pw[l15 * 72 + kk * 32];
            pa.q[0] = *(const uint4*)(pr + half * 8);
            pa.q[1] = *(const uint4*)(pr + 16 + half * 8);
            for (int dt = 0; dt < 4; ++dt) {
                Frag bv;
                const unsigned short* vr = &Vt[(dt * 16 + l15) * 72 + kk * 32 + half * 8];
                bv.q[0] = *(const uint4*)vr;
                bv.q[1] = *(const uint4*)(vr + 16);
                o_acc[dt] = wmma_bf16(pa.v, bv.v, o_acc[dt]);
            }
        }
        __syncthreads();
    }

    // normalize and store bf16 output
    for (int r = 0; r < 8; ++r) {
        float inv = 1.0f / l_run[r];
        int row = q0 + wave * 16 + r + half * 8;
        for (int dt = 0; dt < 4; ++dt)
            O[(size_t)row * D + co + dt * 16 + l15] = f2bf(o_acc[dt][r] * inv);
    }
}

// ---------------------------------------------------------------------------
// Host orchestration
// ---------------------------------------------------------------------------
extern "C" void kernel_launch(void* const* d_in, const int* in_sizes, int n_in,
                              void* d_out, int out_size, void* d_ws, size_t ws_size,
                              hipStream_t stream) {
    const float*     x    = (const float*)d_in[0];
    const long long* eidx = (const long long*)d_in[1];
    const float* W_gcn = (const float*)d_in[2];
    const float* b_gcn = (const float*)d_in[3];
    const float* Wq = (const float*)d_in[4];  const float* bq = (const float*)d_in[5];
    const float* Wk = (const float*)d_in[6];  const float* bk = (const float*)d_in[7];
    const float* Wv = (const float*)d_in[8];  const float* bv = (const float*)d_in[9];
    const float* Wo = (const float*)d_in[10]; const float* bo = (const float*)d_in[11];
    const float* ln1_g = (const float*)d_in[12]; const float* ln1_b = (const float*)d_in[13];
    const float* ln2_g = (const float*)d_in[14]; const float* ln2_b = (const float*)d_in[15];
    const float* ln3_g = (const float*)d_in[16]; const float* ln3_b = (const float*)d_in[17];
    const float* W1 = (const float*)d_in[18]; const float* bf1 = (const float*)d_in[19];
    const float* W2 = (const float*)d_in[20]; const float* bf2 = (const float*)d_in[21];

    const int D = 256;
    const int N = in_sizes[0] / D;
    const long long E = in_sizes[1] / 2;
    float* out = (float*)d_out;

    // ---- workspace carve-up (256B aligned) ----
    char* cur = (char*)d_ws;
    auto alloc = [&](size_t bytes) -> char* {
        char* p = cur; cur += (bytes + 255) & ~(size_t)255; return p;
    };
    unsigned short* xbf   = (unsigned short*)alloc((size_t)N * D * 2);
    unsigned short* wgcnt = (unsigned short*)alloc((size_t)D * D * 2);      // [D][D] transposed
    unsigned short* wqt   = (unsigned short*)alloc((size_t)D * D * 2);
    unsigned short* wkt   = (unsigned short*)alloc((size_t)D * D * 2);
    unsigned short* wvt   = (unsigned short*)alloc((size_t)D * D * 2);
    unsigned short* wot   = (unsigned short*)alloc((size_t)D * D * 2);
    unsigned short* w1t   = (unsigned short*)alloc((size_t)D * 2 * D * 2);  // [2D][D]
    unsigned short* w2t   = (unsigned short*)alloc((size_t)2 * D * D * 2);  // [D][2D]
    float* xw   = (float*)alloc((size_t)N * D * 4);
    float* deg  = (float*)alloc((size_t)N * 4);
    float* dinv = (float*)alloc((size_t)N * 4);
    float* agg  = (float*)alloc((size_t)N * D * 4);
    float* x1   = (float*)alloc((size_t)N * D * 4);
    unsigned short* x1b = (unsigned short*)alloc((size_t)N * D * 2);
    unsigned short* qb  = (unsigned short*)alloc((size_t)N * D * 2);
    unsigned short* kb  = (unsigned short*)alloc((size_t)N * D * 2);
    unsigned short* vtG = (unsigned short*)alloc((size_t)D * N * 2);        // [256][N]
    unsigned short* ob  = (unsigned short*)alloc((size_t)N * D * 2);
    float* t1   = (float*)alloc((size_t)N * D * 4);
    float* x2   = (float*)alloc((size_t)N * D * 4);
    unsigned short* x2b = (unsigned short*)alloc((size_t)N * D * 2);
    unsigned short* hb  = (unsigned short*)alloc((size_t)N * 2 * D * 2);
    float* ffn  = (float*)alloc((size_t)N * D * 4);

    auto cvt = [&](const float* s, unsigned short* d, long long n) {
        int blocks = (int)((n + 255) / 256);
        if (blocks > 4096) blocks = 4096;
        cvt_bf16_k<<<blocks, 256, 0, stream>>>(s, d, n);
    };
    auto cvt_t = [&](const float* s, unsigned short* d, int K, int Nc) {
        long long n = (long long)K * Nc;
        int blocks = (int)((n + 255) / 256);
        if (blocks > 4096) blocks = 4096;
        cvt_t_bf16_k<<<blocks, 256, 0, stream>>>(s, d, K, Nc);
    };
    auto gemm = [&](const unsigned short* A, const unsigned short* Bt, const float* bias,
                    float* C, unsigned short* Cbf, int M, int Nc, int K, int relu, int tN) {
        dim3 grid(Nc / 128, M / 128);
        gemm_bf16_k<<<grid, 256, 0, stream>>>(A, Bt, bias, C, Cbf, M, Nc, K, relu, tN);
    };

    // 1) bf16 conversions (weights transposed once; reused by all row-blocks)
    cvt(x, xbf, (long long)N * D);
    cvt_t(W_gcn, wgcnt, D, D);
    cvt_t(Wq, wqt, D, D);
    cvt_t(Wk, wkt, D, D);
    cvt_t(Wv, wvt, D, D);
    cvt_t(Wo, wot, D, D);
    cvt_t(W1, w1t, D, 2 * D);
    cvt_t(W2, w2t, 2 * D, D);

    // 2) xw = x @ W_gcn
    gemm(xbf, wgcnt, nullptr, xw, nullptr, N, D, D, 0, 0);

    // 3) GCN normalization + scatter
    fill_f32_k<<<256, 256, 0, stream>>>(deg, 1.0f, N);   // self-loop contributes 1
    hipMemsetAsync(agg, 0, (size_t)N * D * 4, stream);
    deg_count_k<<<(int)((E + 255) / 256), 256, 0, stream>>>(eidx, E, deg);
    dinv_k<<<(N + 255) / 256, 256, 0, stream>>>(deg, dinv, N);
    scatter_k<<<(int)E, 256, 0, stream>>>(eidx, E, dinv, xw, agg);

    // 4) x1 = LN(x + local)
    gcn_ln_k<<<N, 256, 0, stream>>>(x, agg, xw, dinv, b_gcn, ln1_g, ln1_b, x1, x1b);

    // 5) Q, K, V projections (V written transposed for flash B-operand)
    gemm(x1b, wqt, bq, nullptr, qb, N, D, D, 0, 0);
    gemm(x1b, wkt, bk, nullptr, kb, N, D, D, 0, 0);
    gemm(x1b, wvt, bv, nullptr, vtG, N, D, D, 0, N);

    // 6) flash attention -> o (bf16)
    {
        dim3 grid(N / 64, 4);
        flash_k<<<grid, 128, 0, stream>>>(qb, kb, vtG, ob, N);
    }

    // 7) o @ Wo + bo ; x2 = LN(x1 + .)
    gemm(ob, wot, bo, t1, nullptr, N, D, D, 0, 0);
    add_ln_k<<<N, 256, 0, stream>>>(x1, t1, ln2_g, ln2_b, x2, x2b);

    // 8) FFN: h = relu(x2 @ W1 + bf1) ; f = h @ W2 + bf2
    gemm(x2b, w1t, bf1, nullptr, hb, N, 2 * D, D, 1, 0);
    gemm(hb, w2t, bf2, ffn, nullptr, N, D, 2 * D, 0, 0);

    // 9) out = LN(x2 + f)
    add_ln_k<<<N, 256, 0, stream>>>(x2, ffn, ln3_g, ln3_b, out, nullptr);
}
